// DecoderLayer_23338852286670
// MI455X (gfx1250) — compile-verified, matrix operand order checked by reading, not run on hardware
//
#include <hip/hip_runtime.h>
#include <cstdint>
#include <cstddef>

#define DEVI __device__ __forceinline__

typedef __attribute__((ext_vector_type(16))) __bf16 v16bf;
typedef __attribute__((ext_vector_type(8)))  __bf16 v8bf;
typedef __attribute__((ext_vector_type(8)))  float  v8f;

// ---------------- problem sizes ----------------
static constexpr int S_    = 4096;
static constexpr int N_KVc = 2048;
static constexpr int N_AUXc= 512;
static constexpr int N_NEWc= 1536;
static constexpr int Hc    = 2048;
static constexpr int NHc   = 16;
static constexpr int HDc   = 128;
static constexpr int FFc   = 8192;
static constexpr int NQc   = N_NEWc + N_AUXc;   // 2048
static constexpr int NKc   = N_KVc + NQc;       // 4096
static constexpr int PRUNEK= 409;
static constexpr float EPSc  = 1e-6f;
static constexpr float SCALEc= 0.08838834764831845f; // 1/sqrt(128)

// ---------------- CDNA5 async-copy helpers (ISA 15.18 / ASYNCcnt) ----------------
DEVI uint32_t lds_addr_of(const void* p) {
  // LDS aperture: flat addr[31:0] is the LDS byte offset (ISA 10.2 aperture table)
  return (uint32_t)(uintptr_t)p;
}
DEVI void async_ld_b128(uint32_t lds_off, const void* gptr) {
  asm volatile("global_load_async_to_lds_b128 %0, %1, off"
               :: "v"(lds_off), "v"((uint64_t)(uintptr_t)gptr)
               : "memory");
}
DEVI void wait_async0() {
  asm volatile("s_wait_asynccnt 0x0" ::: "memory");
}

// ---------------- WMMA helpers ----------------
DEVI v8f wmma_bf16(v16bf a, v16bf b, v8f c) {
  return __builtin_amdgcn_wmma_f32_16x16x32_bf16(false, a, false, b, (short)0, c, false, false);
}

// Load one 16x32 bf16 A-style fragment (also used for B via the [N,K] transposed view).
// lane<16 : row=row0+lane,    K in {0..7, 16..23}
// lane>=16: row=row0+lane-16, K in {8..15, 24..31}
DEVI v16bf load_frag(const __bf16* base, int ld, int row0, int k0) {
  int lane = threadIdx.x & 31;
  int r    = row0 + (lane & 15);
  int koff = (lane >> 4) << 3;
  const __bf16* p = base + (size_t)r * ld + k0 + koff;
  v8bf lo = *(const v8bf*)(p);
  v8bf hi = *(const v8bf*)(p + 16);
  v16bf out;
#pragma unroll
  for (int i = 0; i < 8; ++i) { out[i] = lo[i]; out[8 + i] = hi[i]; }
  return out;
}

DEVI float half_max16(float x) {
#pragma unroll
  for (int m = 1; m < 16; m <<= 1) x = fmaxf(x, __shfl_xor(x, m, 32));
  return x;
}
DEVI float half_sum16(float x) {
#pragma unroll
  for (int m = 1; m < 16; m <<= 1) x += __shfl_xor(x, m, 32);
  return x;
}

// ---------------- small utility kernels ----------------
__global__ void k_init_maps(int* map_hs, int* map_full, int* aux_bits, int* pruned_bits) {
  int i = blockIdx.x * blockDim.x + threadIdx.x;
  if (i < S_) { map_hs[i] = -1; map_full[i] = 0; aux_bits[i] = 0; pruned_bits[i] = 0; }
}

DEVI int hs_index(int i, const int* new_idxs, const int* aux_idxs) {
  return (i < N_NEWc) ? new_idxs[i] : aux_idxs[i - N_NEWc];
}

__global__ void k_build_idx(const int* kv_idxs, const int* aux_idxs, const int* new_idxs,
                            const int* positions, int* hs_idxs, int* key_tok,
                            int* q_pos, int* k_pos, int* map_hs, int* map_full,
                            int* aux_bits, int* lastinfo) {
  int j = blockIdx.x * blockDim.x + threadIdx.x;
  if (j < NKc) {
    int tok = (j < N_KVc) ? kv_idxs[j] : hs_index(j - N_KVc, new_idxs, aux_idxs);
    key_tok[j] = tok;
    k_pos[j]   = positions[tok];
  }
  if (j < NQc) {
    int hv = hs_index(j, new_idxs, aux_idxs);
    hs_idxs[j] = hv;
    q_pos[j]   = positions[hv];
    map_hs[hv]   = j;
    map_full[hv] = j;
    if (hv == S_ - 1) {
      lastinfo[0] = j;                                          // last_hs
      lastinfo[1] = N_KVc + j;                                  // last_key
      lastinfo[2] = (N_KVc + j < NQc) ? (N_KVc + j) : (NQc - 1);// clipped query row
    }
  }
  if (j < N_AUXc) aux_bits[aux_idxs[j]] = 1;
}

__global__ void k_build_hs(const float* hidden, const float* aux_cache,
                           const int* aux_idxs, float* hs) {
  int idx = blockIdx.x * blockDim.x + threadIdx.x;
  if (idx >= NQc * Hc) return;
  int i = idx / Hc, c = idx - i * Hc;
  float v;
  if (i < N_NEWc) v = hidden[(size_t)i * Hc + c];
  else            v = aux_cache[(size_t)aux_idxs[i - N_NEWc] * Hc + c];
  hs[idx] = v;
}

// transpose + convert: src fp32 [K,N] row-major  ->  dst bf16 [N,K]
__global__ void k_convT(const float* src, __bf16* dst, int K, int N) {
  size_t idx = (size_t)blockIdx.x * blockDim.x + threadIdx.x;
  if (idx >= (size_t)K * N) return;
  int k = (int)(idx / N), n = (int)(idx - (size_t)k * N);
  dst[(size_t)n * K + k] = (__bf16)src[idx];
}

__global__ void k_rmsnorm_bf16(const float* x, const float* w, __bf16* out, int rows) {
  __shared__ float red[256];
  int row = blockIdx.x;
  if (row >= rows) return;
  const float* xr = x + (size_t)row * Hc;
  float acc = 0.f;
  for (int c = threadIdx.x; c < Hc; c += 256) { float v = xr[c]; acc += v * v; }
  red[threadIdx.x] = acc; __syncthreads();
  for (int s = 128; s > 0; s >>= 1) { if (threadIdx.x < s) red[threadIdx.x] += red[threadIdx.x + s]; __syncthreads(); }
  float scale = rsqrtf(red[0] / (float)Hc + EPSc);
  for (int c = threadIdx.x; c < Hc; c += 256)
    out[(size_t)row * Hc + c] = (__bf16)(xr[c] * scale * w[c]);
}

__global__ void k_rope(float* qbuf, float* kbuf, const int* q_pos) {
  int idx = blockIdx.x * blockDim.x + threadIdx.x;
  if (idx >= NQc * NHc * (HDc / 2)) return;
  int i = idx / (NHc * 64);
  int r = idx - i * (NHc * 64);
  int h = r / 64, d = r - h * 64;
  float pos  = (float)q_pos[i];
  float freq = __powf(10000.0f, -(float)(2 * d) / (float)HDc);
  float a = pos * freq, ca = cosf(a), sa = sinf(a);
  size_t b = (size_t)i * Hc + h * HDc + d;
  float qx = qbuf[b], qy = qbuf[b + 64];
  qbuf[b] = qx * ca - qy * sa; qbuf[b + 64] = qy * ca + qx * sa;
  float kx = kbuf[b], ky = kbuf[b + 64];
  kbuf[b] = kx * ca - ky * sa; kbuf[b + 64] = ky * ca + kx * sa;
}

__global__ void k_cvt_bf16(const float* src, __bf16* dst, int n) {
  int i = blockIdx.x * blockDim.x + threadIdx.x;
  if (i < n) dst[i] = (__bf16)src[i];
}

// gather K/V (bf16): kb [NH][NK][HD], vbT [NH][HD][NK]
__global__ void k_gather_kv(const float* key_cache, const float* value_cache,
                            const int* kv_idxs, const float* kbuf, const float* vbuf,
                            __bf16* kb, __bf16* vbT) {
  int idx = blockIdx.x * blockDim.x + threadIdx.x;
  if (idx >= NHc * NKc * HDc) return;
  int h = idx / (NKc * HDc);
  int r = idx - h * (NKc * HDc);
  int j = r / HDc, d = r - j * HDc;
  float kv, vv;
  if (j < N_KVc) {
    int s = kv_idxs[j];
    kv = key_cache  [((size_t)h * S_ + s) * HDc + d];
    vv = value_cache[((size_t)h * S_ + s) * HDc + d];
  } else {
    int m = j - N_KVc;
    kv = kbuf[(size_t)m * Hc + h * HDc + d];
    vv = vbuf[(size_t)m * Hc + h * HDc + d];
  }
  kb [(size_t)h * NKc * HDc + (size_t)j * HDc + d] = (__bf16)kv;
  vbT[(size_t)h * HDc * NKc + (size_t)d * NKc + j] = (__bf16)vv;
}

// ---------------- GEMM: C[M,N] = A[M,K](bf16) x BT[N,K](bf16) ----------------
// 256 threads (8 waves), 128x128 block tile, K-panel 64, double-buffered LDS
// filled via global_load_async_to_lds_b128 (ASYNCcnt path).
__global__ __launch_bounds__(256)
void k_gemm(const __bf16* A, const __bf16* BT, int M, int N, int K,
            float* Cf, __bf16* Cb, const float* res, const float* gate) {
  __shared__ __align__(16) __bf16 shA[2][128 * 64];
  __shared__ __align__(16) __bf16 shB[2][128 * 64];
  int tid  = threadIdx.x;
  int wave = tid >> 5;
  int wm = wave & 3, wn = wave >> 2;       // 4x2 wave grid
  int m0 = wm * 32, n0 = wn * 64;          // wave tile 32x64 inside block tile
  int bm = blockIdx.y * 128, bn = blockIdx.x * 128;
  const int KB = 64;
  int nkb = K / KB;

  auto issue = [&](int buf, int kb) {
    uint32_t la = lds_addr_of(&shA[buf][0]);
    uint32_t lb = lds_addr_of(&shB[buf][0]);
    int kbyte = kb * KB * 2;
#pragma unroll
    for (int i = 0; i < 4; ++i) {
      int c = tid + 256 * i;               // 0..1023 chunks of 16B
      int row = c >> 3, cw = c & 7;
      const char* ga = (const char*)(A  + (size_t)(bm + row) * K) + kbyte + cw * 16;
      const char* gb = (const char*)(BT + (size_t)(bn + row) * K) + kbyte + cw * 16;
      async_ld_b128(la + row * 128 + cw * 16, ga);
      async_ld_b128(lb + row * 128 + cw * 16, gb);
    }
  };

  v8f acc[2][4] = {};
  issue(0, 0);
  for (int kb = 0; kb < nkb; ++kb) {
    int cur = kb & 1;
    wait_async0();
    __syncthreads();
    if (kb + 1 < nkb) issue(cur ^ 1, kb + 1);
#pragma unroll
    for (int ks = 0; ks < 2; ++ks) {
      int k0 = ks * 32;
      v16bf af[2], bfr[4];
#pragma unroll
      for (int mi = 0; mi < 2; ++mi) af[mi]  = load_frag(&shA[cur][0], KB, m0 + mi * 16, k0);
#pragma unroll
      for (int ni = 0; ni < 4; ++ni) bfr[ni] = load_frag(&shB[cur][0], KB, n0 + ni * 16, k0);
#pragma unroll
      for (int mi = 0; mi < 2; ++mi)
#pragma unroll
        for (int ni = 0; ni < 4; ++ni)
          acc[mi][ni] = wmma_bf16(af[mi], bfr[ni], acc[mi][ni]);
    }
    __syncthreads();
  }

  int lane = tid & 31, col = lane & 15, mb = (lane >> 4) * 8;
#pragma unroll
  for (int mi = 0; mi < 2; ++mi)
#pragma unroll
    for (int ni = 0; ni < 4; ++ni)
#pragma unroll
      for (int j = 0; j < 8; ++j) {
        int m = bm + m0 + mi * 16 + mb + j;
        int n = bn + n0 + ni * 16 + col;
        size_t idx = (size_t)m * N + n;
        float c = acc[mi][ni][j];
        if (gate) { float g = gate[idx]; c = (g / (1.f + __expf(-g))) * c; }
        if (res)  c += res[idx];
        if (Cf) Cf[idx] = c;
        if (Cb) Cb[idx] = (__bf16)c;
      }
}

// ---------------- flash attention ----------------
// 128 threads (4 waves), one head per blockIdx.y, 64 query rows per block
// (16 per wave). K/V tiles (32 keys) double-buffered in LDS via async copies
// and shared by all 4 waves.
__global__ __launch_bounds__(128)
void k_flash(const __bf16* qb, const __bf16* kb, const __bf16* vbT,
             const int* q_pos, const int* k_pos, __bf16* ctxb) {
  __shared__ __align__(16) __bf16 shK[2][32 * 128];  // [key][d]  ld=128
  __shared__ __align__(16) __bf16 shV[2][128 * 32];  // [d][key]  ld=32
  __shared__ __align__(16) __bf16 pls[4][16 * 32];
  int h    = blockIdx.y;
  int tid  = threadIdx.x;
  int wave = tid >> 5;
  int lane = tid & 31;
  int col = lane & 15, mb = (lane >> 4) * 8;
  int qt = blockIdx.x * 4 + wave;                    // 16-row query tile
  const __bf16* kbh = kb  + (size_t)h * NKc * HDc;
  const __bf16* vbh = vbT + (size_t)h * HDc * NKc;

  auto issue = [&](int buf, int key0) {
    uint32_t lk = lds_addr_of(&shK[buf][0]);
    uint32_t lv = lds_addr_of(&shV[buf][0]);
#pragma unroll
    for (int i = 0; i < 4; ++i) {
      int c = tid + 128 * i;                          // 0..511
      int kr = c >> 4, kc = c & 15;                   // K tile: 32 rows x 256B
      async_ld_b128(lk + kr * 256 + kc * 16,
                    (const char*)kbh + (size_t)(key0 + kr) * 256 + kc * 16);
      int vr = c >> 2, vc = c & 3;                    // V tile: 128 rows x 64B
      async_ld_b128(lv + vr * 64 + vc * 16,
                    (const char*)vbh + (size_t)vr * (NKc * 2) + key0 * 2 + vc * 16);
    }
  };

  v16bf aq[4];
#pragma unroll
  for (int kk = 0; kk < 4; ++kk)
    aq[kk] = load_frag(qb, Hc, qt * 16, h * HDc + kk * 32);

  int qp[8];
#pragma unroll
  for (int j = 0; j < 8; ++j) qp[j] = q_pos[qt * 16 + mb + j];

  v8f zero = {0.f, 0.f, 0.f, 0.f, 0.f, 0.f, 0.f, 0.f};
  v8f o[8];
#pragma unroll
  for (int t = 0; t < 8; ++t) o[t] = zero;
  float mrow[8], lrow[8];
#pragma unroll
  for (int j = 0; j < 8; ++j) { mrow[j] = -3.0e38f; lrow[j] = 0.f; }

  issue(0, 0);
  for (int key0 = 0; key0 < NKc; key0 += 32) {
    int cur = (key0 >> 5) & 1;
    wait_async0();
    __syncthreads();
    if (key0 + 32 < NKc) issue(cur ^ 1, key0 + 32);

    v8f s0 = zero, s1 = zero;
#pragma unroll
    for (int kk = 0; kk < 4; ++kk) {
      v16bf b0 = load_frag(&shK[cur][0], HDc, 0,  kk * 32);
      v16bf b1 = load_frag(&shK[cur][0], HDc, 16, kk * 32);
      s0 = wmma_bf16(aq[kk], b0, s0);
      s1 = wmma_bf16(aq[kk], b1, s1);
    }
    int kp0 = k_pos[key0 + col];
    int kp1 = k_pos[key0 + 16 + col];
    float corr8[8];
#pragma unroll
    for (int j = 0; j < 8; ++j) {
      float v0 = s0[j] * SCALEc + ((kp0 <= qp[j]) ? 0.f : -1e9f);
      float v1 = s1[j] * SCALEc + ((kp1 <= qp[j]) ? 0.f : -1e9f);
      float tm = half_max16(fmaxf(v0, v1));
      float nm = fmaxf(mrow[j], tm);
      float corr = __expf(mrow[j] - nm);
      float p0 = __expf(v0 - nm);
      float p1 = __expf(v1 - nm);
      float rs = half_sum16(p0 + p1);
      lrow[j] = lrow[j] * corr + rs;
      mrow[j] = nm;
      corr8[j] = corr;
      pls[wave][(mb + j) * 32 + col]      = (__bf16)p0;
      pls[wave][(mb + j) * 32 + 16 + col] = (__bf16)p1;
    }
#pragma unroll
    for (int t = 0; t < 8; ++t)
#pragma unroll
      for (int j = 0; j < 8; ++j) o[t][j] *= corr8[j];
    __syncthreads();
    v16bf ap = load_frag(&pls[wave][0], 32, 0, 0);
#pragma unroll
    for (int t = 0; t < 8; ++t) {
      v16bf bv = load_frag(&shV[cur][0], 32, t * 16, 0);
      o[t] = wmma_bf16(ap, bv, o[t]);
    }
    __syncthreads();
  }
#pragma unroll
  for (int t = 0; t < 8; ++t)
#pragma unroll
    for (int j = 0; j < 8; ++j) {
      int row = qt * 16 + mb + j;
      float val = o[t][j] / lrow[j];
      ctxb[(size_t)row * Hc + h * HDc + t * 16 + col] = (__bf16)val;
    }
}

// ---------------- importance row (per head) ----------------
__global__ __launch_bounds__(256)
void k_importance(const __bf16* qb, const __bf16* kb, const int* q_pos, const int* k_pos,
                  const int* lastinfo, float* impH) {
  __shared__ float sc[NKc];
  __shared__ float qs[HDc];
  __shared__ float red[256];
  int h = blockIdx.x;
  int qi = lastinfo[2];
  if (threadIdx.x < HDc) qs[threadIdx.x] = (float)qb[(size_t)qi * Hc + h * HDc + threadIdx.x];
  __syncthreads();
  int qpv = q_pos[qi];
  const __bf16* kbh = kb + (size_t)h * NKc * HDc;
  float lmax = -3.0e38f;
  for (int k = threadIdx.x; k < NKc; k += 256) {
    float acc = 0.f;
    const __bf16* kr = kbh + (size_t)k * HDc;
    for (int d = 0; d < HDc; ++d) acc += qs[d] * (float)kr[d];
    float s = acc * SCALEc + ((k_pos[k] <= qpv) ? 0.f : -1e9f);
    sc[k] = s;
    lmax = fmaxf(lmax, s);
  }
  red[threadIdx.x] = lmax; __syncthreads();
  for (int s = 128; s > 0; s >>= 1) { if (threadIdx.x < s) red[threadIdx.x] = fmaxf(red[threadIdx.x], red[threadIdx.x + s]); __syncthreads(); }
  float mx = red[0]; __syncthreads();
  float lsum = 0.f;
  for (int k = threadIdx.x; k < NKc; k += 256) lsum += __expf(sc[k] - mx);
  red[threadIdx.x] = lsum; __syncthreads();
  for (int s = 128; s > 0; s >>= 1) { if (threadIdx.x < s) red[threadIdx.x] += red[threadIdx.x + s]; __syncthreads(); }
  float inv = 1.f / red[0];
  for (int k = threadIdx.x; k < NKc; k += 256)
    impH[(size_t)h * NKc + k] = __expf(sc[k] - mx) * inv * (1.0f / (float)NHc);
}

// ---------------- top-k smallest -> pruned token bitmap ----------------
__global__ __launch_bounds__(256)
void k_topk(const float* impH, const int* key_tok, const int* lastinfo, int* pruned_bits) {
  __shared__ float v[NKc];
  __shared__ float bv[256];
  __shared__ int   bi[256];
  for (int k = threadIdx.x; k < NKc; k += 256) {
    float acc = 0.f;
    for (int h = 0; h < NHc; ++h) acc += impH[(size_t)h * NKc + k];
    v[k] = acc;
  }
  __syncthreads();
  if (threadIdx.x == 0) v[lastinfo[1]] = 3.0e38f;
  __syncthreads();
  for (int it = 0; it < PRUNEK; ++it) {
    float bestv = 3.0e38f; int besti = NKc;
    for (int k = threadIdx.x; k < NKc; k += 256) {
      float x = v[k];
      if (x < bestv || (x == bestv && k < besti)) { bestv = x; besti = k; }
    }
    bv[threadIdx.x] = bestv; bi[threadIdx.x] = besti; __syncthreads();
    for (int s = 128; s > 0; s >>= 1) {
      if (threadIdx.x < s) {
        float ov = bv[threadIdx.x + s]; int oi = bi[threadIdx.x + s];
        if (ov < bv[threadIdx.x] || (ov == bv[threadIdx.x] && oi < bi[threadIdx.x])) {
          bv[threadIdx.x] = ov; bi[threadIdx.x] = oi;
        }
      }
      __syncthreads();
    }
    if (threadIdx.x == 0) {
      int idx = bi[0];
      pruned_bits[key_tok[idx]] = 1;
      v[idx] = 3.0e38f;
    }
    __syncthreads();
  }
}

// ---------------- output scatter kernels ----------------
__global__ void k_kv_out(const float* key_cache, const float* value_cache,
                         const float* kbuf, const float* vbuf, const int* map_hs,
                         float* kout, float* vout) {
  int idx = blockIdx.x * blockDim.x + threadIdx.x;
  if (idx >= NHc * S_ * HDc) return;
  int h = idx / (S_ * HDc);
  int r = idx - h * (S_ * HDc);
  int s = r / HDc, d = r - s * HDc;
  int m = map_hs[s];
  float kv, vv;
  if (m >= 0) {
    kv = kbuf[(size_t)m * Hc + h * HDc + d];
    vv = vbuf[(size_t)m * Hc + h * HDc + d];
  } else {
    kv = key_cache[idx];
    vv = value_cache[idx];
  }
  kout[idx] = kv;
  vout[idx] = vv;
}

__global__ void k_aux_out(const float* aux_cache, const float* outbuf,
                          const int* pruned_bits, const int* aux_bits,
                          const int* map_full, float* aux_new) {
  int idx = blockIdx.x * blockDim.x + threadIdx.x;
  if (idx >= S_ * Hc) return;
  int s = idx / Hc, c = idx - s * Hc;
  bool add = pruned_bits[s] && !aux_bits[s];
  float v = add ? outbuf[(size_t)map_full[s] * Hc + c] : aux_cache[idx];
  aux_new[idx] = v;
}

// ---------------- host orchestration ----------------
extern "C" void kernel_launch(void* const* d_in, const int* in_sizes, int n_in,
                              void* d_out, int out_size, void* d_ws, size_t ws_size,
                              hipStream_t stream) {
  (void)in_sizes; (void)n_in; (void)out_size; (void)ws_size;
  const float* hidden      = (const float*)d_in[0];
  const float* key_cache   = (const float*)d_in[1];
  const float* value_cache = (const float*)d_in[2];
  const float* aux_cache   = (const float*)d_in[3];
  const int*   kv_idxs     = (const int*)d_in[4];
  const int*   aux_idxs    = (const int*)d_in[5];
  const int*   new_idxs    = (const int*)d_in[6];
  const int*   positions   = (const int*)d_in[7];
  const float* wq = (const float*)d_in[8];
  const float* wk = (const float*)d_in[9];
  const float* wv = (const float*)d_in[10];
  const float* wo = (const float*)d_in[11];
  const float* wg = (const float*)d_in[12];
  const float* wu = (const float*)d_in[13];
  const float* wd = (const float*)d_in[14];
  const float* ln1 = (const float*)d_in[15];
  const float* ln2 = (const float*)d_in[16];

  char* ws = (char*)d_ws;
  size_t off = 0;
  auto alloc = [&](size_t bytes) -> char* {
    char* p = ws + off;
    off = (off + bytes + 255) & ~(size_t)255;
    return p;
  };
  __bf16* wqT = (__bf16*)alloc((size_t)Hc * Hc * 2);
  __bf16* wkT = (__bf16*)alloc((size_t)Hc * Hc * 2);
  __bf16* wvT = (__bf16*)alloc((size_t)Hc * Hc * 2);
  __bf16* woT = (__bf16*)alloc((size_t)Hc * Hc * 2);
  __bf16* wgT = (__bf16*)alloc((size_t)FFc * Hc * 2);
  __bf16* wuT = (__bf16*)alloc((size_t)FFc * Hc * 2);
  __bf16* wdT = (__bf16*)alloc((size_t)Hc * FFc * 2);
  float*  hs    = (float*) alloc((size_t)NQc * Hc * 4);
  __bf16* xnb   = (__bf16*)alloc((size_t)NQc * Hc * 2);
  float*  qbuf  = (float*) alloc((size_t)NQc * Hc * 4);
  float*  kbuf  = (float*) alloc((size_t)NQc * Hc * 4);
  float*  vbuf  = (float*) alloc((size_t)NQc * Hc * 4);
  __bf16* qb    = (__bf16*)alloc((size_t)NQc * Hc * 2);
  __bf16* kb    = (__bf16*)alloc((size_t)NHc * NKc * HDc * 2);
  __bf16* vbT   = (__bf16*)alloc((size_t)NHc * HDc * NKc * 2);
  __bf16* ctxb  = (__bf16*)alloc((size_t)NQc * Hc * 2);
  float*  hs2   = (float*) alloc((size_t)NQc * Hc * 4);
  __bf16* yb    = (__bf16*)alloc((size_t)NQc * Hc * 2);
  float*  g     = (float*) alloc((size_t)NQc * FFc * 4);
  __bf16* actb  = (__bf16*)alloc((size_t)NQc * FFc * 2);
  float*  impH  = (float*) alloc((size_t)NHc * NKc * 4);
  int* hs_idxs  = (int*)alloc(NQc * 4);
  int* key_tok  = (int*)alloc(NKc * 4);
  int* q_pos    = (int*)alloc(NQc * 4);
  int* k_pos    = (int*)alloc(NKc * 4);
  int* map_hs   = (int*)alloc(S_ * 4);
  int* map_full = (int*)alloc(S_ * 4);
  int* aux_bits = (int*)alloc(S_ * 4);
  int* pruned   = (int*)alloc(S_ * 4);
  int* lastinfo = (int*)alloc(16);

  float* outbuf = (float*)d_out;                                   // [NQ,H]
  float* kout   = outbuf + (size_t)NQc * Hc;                       // [NH,S,HD]
  float* vout   = kout   + (size_t)NHc * S_ * HDc;
  float* auxout = vout   + (size_t)NHc * S_ * HDc;                 // [S,H]

  auto blk = [](size_t n) { return dim3((unsigned)((n + 255) / 256)); };

  // 1) maps/indices
  k_init_maps<<<blk(S_), 256, 0, stream>>>(map_hs, map_full, aux_bits, pruned);
  k_build_idx<<<blk(NKc), 256, 0, stream>>>(kv_idxs, aux_idxs, new_idxs, positions,
                                            hs_idxs, key_tok, q_pos, k_pos,
                                            map_hs, map_full, aux_bits, lastinfo);
  // 2) weight convert+transpose (bf16, K-contiguous)
  k_convT<<<blk((size_t)Hc * Hc), 256, 0, stream>>>(wq, wqT, Hc, Hc);
  k_convT<<<blk((size_t)Hc * Hc), 256, 0, stream>>>(wk, wkT, Hc, Hc);
  k_convT<<<blk((size_t)Hc * Hc), 256, 0, stream>>>(wv, wvT, Hc, Hc);
  k_convT<<<blk((size_t)Hc * Hc), 256, 0, stream>>>(wo, woT, Hc, Hc);
  k_convT<<<blk((size_t)Hc * FFc), 256, 0, stream>>>(wg, wgT, Hc, FFc);
  k_convT<<<blk((size_t)Hc * FFc), 256, 0, stream>>>(wu, wuT, Hc, FFc);
  k_convT<<<blk((size_t)FFc * Hc), 256, 0, stream>>>(wd, wdT, FFc, Hc);
  // 3) hs concat + rmsnorm1
  k_build_hs<<<blk((size_t)NQc * Hc), 256, 0, stream>>>(hidden, aux_cache, aux_idxs, hs);
  k_rmsnorm_bf16<<<NQc, 256, 0, stream>>>(hs, ln1, xnb, NQc);
  // 4) QKV projections (WMMA bf16, async-LDS staged)
  dim3 gHH(Hc / 128, NQc / 128), gFF(FFc / 128, NQc / 128);
  k_gemm<<<gHH, 256, 0, stream>>>(xnb, wqT, NQc, Hc, Hc, qbuf, nullptr, nullptr, nullptr);
  k_gemm<<<gHH, 256, 0, stream>>>(xnb, wkT, NQc, Hc, Hc, kbuf, nullptr, nullptr, nullptr);
  k_gemm<<<gHH, 256, 0, stream>>>(xnb, wvT, NQc, Hc, Hc, vbuf, nullptr, nullptr, nullptr);
  // 5) RoPE, bf16 staging for attention
  k_rope<<<blk((size_t)NQc * NHc * 64), 256, 0, stream>>>(qbuf, kbuf, q_pos);
  k_cvt_bf16<<<blk((size_t)NQc * Hc), 256, 0, stream>>>(qbuf, qb, NQc * Hc);
  k_gather_kv<<<blk((size_t)NHc * NKc * HDc), 256, 0, stream>>>(key_cache, value_cache,
                                                                kv_idxs, kbuf, vbuf, kb, vbT);
  // 6) flash attention (WMMA, shared async-LDS K/V tiles)
  dim3 gFA(NQc / 64, NHc);
  k_flash<<<gFA, 128, 0, stream>>>(qb, kb, vbT, q_pos, k_pos, ctxb);
  // 7) O-proj + residual, rmsnorm2, FFN
  k_gemm<<<gHH, 256, 0, stream>>>(ctxb, woT, NQc, Hc, Hc, hs2, nullptr, hs, nullptr);
  k_rmsnorm_bf16<<<NQc, 256, 0, stream>>>(hs2, ln2, yb, NQc);
  k_gemm<<<gFF, 256, 0, stream>>>(yb, wgT, NQc, FFc, Hc, g, nullptr, nullptr, nullptr);
  k_gemm<<<gFF, 256, 0, stream>>>(yb, wuT, NQc, FFc, Hc, nullptr, actb, nullptr, g);
  k_gemm<<<gHH, 256, 0, stream>>>(actb, wdT, NQc, Hc, FFc, outbuf, nullptr, hs2, nullptr);
  // 8) importance + prune set
  k_importance<<<NHc, 256, 0, stream>>>(qb, kb, q_pos, k_pos, lastinfo, impH);
  k_topk<<<1, 256, 0, stream>>>(impH, key_tok, lastinfo, pruned);
  // 9) cache + aux outputs
  k_kv_out<<<blk((size_t)NHc * S_ * HDc), 256, 0, stream>>>(key_cache, value_cache,
                                                            kbuf, vbuf, map_hs, kout, vout);
  k_aux_out<<<blk((size_t)S_ * Hc), 256, 0, stream>>>(aux_cache, outbuf, pruned,
                                                      aux_bits, map_full, auxout);
}
